// MLPClassifier_33586644255247
// MI455X (gfx1250) — matrix operand order repaired and checked by reading, not compiled, not run on hardware
//
#include <hip/hip_runtime.h>

#define IN_DIM 12288
#define HID    4096
#define ODIM   6
#define BATCH  128

typedef __attribute__((ext_vector_type(16))) __bf16 v16bf;
typedef __attribute__((ext_vector_type(8)))  float  v8f;

union BFPack { unsigned int u[8]; v16bf v; };

// round-to-nearest-even f32 -> bf16 (finite inputs)
static __device__ __forceinline__ unsigned int bf16_rne(float f) {
  unsigned int u = __float_as_uint(f);
  return (u + 0x7fffu + ((u >> 16) & 1u)) >> 16;
}

// pack two f32 -> two bf16 in one dword (round-half-up):
// v_add_nc_u32 x2 + v_perm_b32 taking high halves (bytes {7,6,3,2} of {hi,lo})
static __device__ __forceinline__ unsigned int pack_bf16_fast(unsigned int flo, unsigned int fhi) {
  return __builtin_amdgcn_perm(fhi + 0x8000u, flo + 0x8000u, 0x07060302u);
}

__global__ __launch_bounds__(256) void convert_x(const float* __restrict__ X,
                                                 unsigned int* __restrict__ Xbf) {
  int i = blockIdx.x * 256 + threadIdx.x;           // dword index = 2 elements
  if (i < BATCH * IN_DIM / 2) {
    float f0 = X[2 * i];
    float f1 = X[2 * i + 1];
    Xbf[i] = (bf16_rne(f1) << 16) | bf16_rne(f0);
  }
}

// One wave per block: 128(M) x 16(N) output strip over one K-chunk.
// grid = (HID/16, nk). Partials P[nk][BATCH][HID]. W1 is read exactly once from HBM.
__global__ __launch_bounds__(32) void gemm1_bf16_wmma(
    const unsigned int* __restrict__ Xbf,   // (BATCH, IN_DIM) bf16, packed pairs
    const float*        __restrict__ W1,    // (IN_DIM, HID) f32 row-major
    float*              __restrict__ P)     // (nk, BATCH, HID) f32 partial sums
{
  const int lane = threadIdx.x;
  const int lo = lane & 15;
  const int hi = lane >> 4;
  const int n0 = blockIdx.x * 16;
  const int kchunk = IN_DIM / gridDim.y;
  const int kbeg = blockIdx.y * kchunk;
  const int kend = kbeg + kchunk;

  v8f acc[8] = {};

  const float* wcol = W1 + n0 + lo;               // this lane's W1 column
  const int xrow_dw = IN_DIM / 2;                 // dwords per bf16 row

  float  braw[16];                                // next B tile, raw f32 (double buffer)
  BFPack a[8];                                    // A tiles for current k (rotating prefetch)

  // ---- prologue: issue loads for kbeg
  {
    const float* wk = wcol + (size_t)(kbeg + 16 * hi) * HID;
    #pragma unroll
    for (int e = 0; e < 16; ++e) braw[e] = wk[(size_t)e * HID];
    const unsigned int* xr = Xbf + lo * xrow_dw + (kbeg >> 1) + 8 * hi;
    #pragma unroll
    for (int t = 0; t < 8; ++t) {
      *(uint4*)&a[t].u[0] = *(const uint4*)(xr + t * 16 * xrow_dw);
      *(uint4*)&a[t].u[4] = *(const uint4*)(xr + t * 16 * xrow_dw + 4);
    }
  }

  for (int k = kbeg; k < kend; k += 32) {
    const int kn = (k + 32 < kend) ? (k + 32) : k;  // last iter: harmless reload

    // ---- convert current B (waits on braw loads issued last iteration)
    BFPack b;
    #pragma unroll
    for (int e = 0; e < 8; ++e)
      b.u[e] = pack_bf16_fast(__float_as_uint(braw[2 * e]),
                              __float_as_uint(braw[2 * e + 1]));

    // ---- issue B loads for next iteration (hidden under this iteration's WMMAs)
    {
      const float* wk = wcol + (size_t)(kn + 16 * hi) * HID;
      #pragma unroll
      for (int e = 0; e < 16; ++e) braw[e] = wk[(size_t)e * HID];
    }

    // ---- stream-prefetch W1 ~8 iterations ahead: 32 lanes cover rows krow..krow+31
    {
      int krow = k + 256;
      if (krow > IN_DIM - 32) krow = IN_DIM - 32;
      __builtin_prefetch(W1 + (size_t)(krow + lane) * HID + n0, 0, 1);
    }

    // ---- 8 WMMAs; after consuming a[t], immediately prefetch its k+32 tile
    const unsigned int* xr = Xbf + lo * xrow_dw + (kn >> 1) + 8 * hi;
    #pragma unroll
    for (int t = 0; t < 8; ++t) {
      acc[t] = __builtin_amdgcn_wmma_f32_16x16x32_bf16(
          /*neg_a=*/false, a[t].v, /*neg_b=*/false, b.v,
          /*c_mod=*/(short)0, acc[t], /*reuse_a=*/false, /*reuse_b=*/false);
      *(uint4*)&a[t].u[0] = *(const uint4*)(xr + t * 16 * xrow_dw);
      *(uint4*)&a[t].u[4] = *(const uint4*)(xr + t * 16 * xrow_dw + 4);
    }
  }

  // ---- epilogue: C layout VGPR r -> row (r + 8*hi), col lo
  float* pout = P + (size_t)blockIdx.y * (BATCH * HID) + n0 + lo;
  #pragma unroll
  for (int t = 0; t < 8; ++t) {
    #pragma unroll
    for (int r = 0; r < 8; ++r) {
      pout[(size_t)(16 * t + r + 8 * hi) * HID] = acc[t][r];
    }
  }
}

// Sum K-split partials + b1, ReLU, multiply by W2 (4096x6), add b2.
// One block (256 threads) per batch row; fixed-order tree reduction -> deterministic.
__global__ __launch_bounds__(256) void layer2(
    const float* __restrict__ P, const float* __restrict__ b1,
    const float* __restrict__ W2, const float* __restrict__ b2,
    float* __restrict__ out, int nk)
{
  __shared__ float red[256];
  const int row = blockIdx.x;
  const int tid = threadIdx.x;
  float acc[ODIM] = {};
  for (int c = tid; c < HID; c += 256) {
    float s = b1[c];
    for (int p = 0; p < nk; ++p)
      s += P[(size_t)p * (BATCH * HID) + (size_t)row * HID + c];
    s = fmaxf(s, 0.0f);
    #pragma unroll
    for (int j = 0; j < ODIM; ++j)
      acc[j] += s * W2[c * ODIM + j];
  }
  #pragma unroll 1
  for (int j = 0; j < ODIM; ++j) {
    red[tid] = acc[j];
    __syncthreads();
    for (int off = 128; off > 0; off >>= 1) {
      if (tid < off) red[tid] += red[tid + off];
      __syncthreads();
    }
    if (tid == 0) out[row * ODIM + j] = red[0] + b2[j];
    __syncthreads();
  }
}

extern "C" void kernel_launch(void* const* d_in, const int* in_sizes, int n_in,
                              void* d_out, int out_size, void* d_ws, size_t ws_size,
                              hipStream_t stream) {
  const float* X  = (const float*)d_in[0];
  const float* W1 = (const float*)d_in[1];
  const float* b1 = (const float*)d_in[2];
  const float* W2 = (const float*)d_in[3];
  const float* b2 = (const float*)d_in[4];
  float* out = (float*)d_out;

  const size_t xbf_bytes = (size_t)BATCH * IN_DIM * sizeof(unsigned short);
  // K-split factor: prefer 8 (2048 waves), shrink if workspace is tight.
  int nk = 8;
  while (nk > 1 && (size_t)nk * BATCH * HID * sizeof(float) + xbf_bytes > ws_size)
    nk >>= 1;

  float* P = (float*)d_ws;
  unsigned int* Xbf = (unsigned int*)((char*)d_ws + (size_t)nk * BATCH * HID * sizeof(float));

  convert_x<<<(BATCH * IN_DIM / 2 + 255) / 256, 256, 0, stream>>>(X, Xbf);
  dim3 g1(HID / 16, nk);
  gemm1_bf16_wmma<<<g1, 32, 0, stream>>>(Xbf, W1, P);
  layer2<<<BATCH, 256, 0, stream>>>(P, b1, W2, b2, out, nk);
}
